// MultiHeadAttention_14714557956492
// MI455X (gfx1250) — matrix operand order repaired and checked by reading, not compiled, not run on hardware
//
#include <hip/hip_runtime.h>
#include <math.h>

// ---------------------------------------------------------------------------
// Problem constants (reference: B=8, S=4096, H=1024, NH=16, DH=64)
// ---------------------------------------------------------------------------
#define BB   8
#define SS   4096
#define HH   1024
#define NHH  16
#define DHH  64
#define MM   (BB * SS)          // 32768 rows

typedef __attribute__((ext_vector_type(16))) __bf16 v16bf;
typedef __attribute__((ext_vector_type(8)))  __bf16 bf16x8;
typedef __attribute__((ext_vector_type(4)))  __bf16 bf16x4;
typedef __attribute__((ext_vector_type(8)))  float  v8f;
typedef __attribute__((ext_vector_type(4)))  float  f32x4;

union BF16x16 { v16bf v; bf16x8 h[2]; };

// ---------------------------------------------------------------------------
// CDNA5 async global->LDS staging (GLOBAL_LOAD_ASYNC_TO_LDS_B128, ASYNCcnt).
// The clang builtin wants a cuda_device-LangAS vector pointer that can't be
// spelled portably in HIP, so use inline asm per the ISA doc (VDST = LDS byte
// offset VGPR, VADDR = 64-bit global addr; IOFFSET applies to both sides).
// Set to 0 to fall back to register staging + ds_store_b128.
// ---------------------------------------------------------------------------
#define USE_ASYNC_LDS 1

#if USE_ASYNC_LDS
static __device__ __forceinline__ void async_b128x2(const __bf16* g, __bf16* l) {
  // LDS byte offset = low 32 bits of the generic pointer (shared aperture)
  unsigned loff = (unsigned)(size_t)l;
  asm volatile(
      "global_load_async_to_lds_b128 %0, %1, off\n\t"
      "global_load_async_to_lds_b128 %0, %1, off offset:16"
      :: "v"(loff), "v"(g) : "memory");
}
static __device__ __forceinline__ void wait_async0() {
  asm volatile("s_wait_asynccnt 0x0" ::: "memory");
}
#endif

static __device__ __forceinline__ __bf16 f2bf(float f) {
  unsigned u = __builtin_bit_cast(unsigned, f);
  unsigned r = u + 0x7FFFu + ((u >> 16) & 1u);     // round-to-nearest-even
  unsigned short s = (unsigned short)(r >> 16);
  return __builtin_bit_cast(__bf16, s);
}

// ---------------------------------------------------------------------------
// fp32 -> bf16 conversion (X: 32768x1024)
// ---------------------------------------------------------------------------
__global__ __launch_bounds__(256) void cvt_f32_bf16_kernel(
    const float* __restrict__ x, __bf16* __restrict__ y, int n4) {
  int i = blockIdx.x * 256 + threadIdx.x;
  if (i >= n4) return;
  f32x4 v = *(const f32x4*)(x + (size_t)i * 4);
  bf16x4 o;
  o[0] = f2bf(v[0]); o[1] = f2bf(v[1]); o[2] = f2bf(v[2]); o[3] = f2bf(v[3]);
  *(bf16x4*)(y + (size_t)i * 4) = o;
}

// ---------------------------------------------------------------------------
// Weight transpose + convert: W[k][n] (fp32) -> Wt[n][k] (bf16), 1024x1024
// ---------------------------------------------------------------------------
__global__ __launch_bounds__(256) void transpose_w_kernel(
    const float* __restrict__ W, __bf16* __restrict__ Wt) {
  __shared__ float tile[32][33];
  int tx = threadIdx.x & 31, ty = threadIdx.x >> 5;
  int n0 = blockIdx.x * 32, k0 = blockIdx.y * 32;
#pragma unroll
  for (int r = 0; r < 4; ++r)
    tile[ty + 8 * r][tx] = W[(size_t)(k0 + ty + 8 * r) * HH + (n0 + tx)];
  __syncthreads();
#pragma unroll
  for (int r = 0; r < 4; ++r)
    Wt[(size_t)(n0 + ty + 8 * r) * HH + (k0 + tx)] = f2bf(tile[tx][ty + 8 * r]);
}

__global__ __launch_bounds__(256) void zero_f32_kernel(float* __restrict__ p, int n) {
  int i = blockIdx.x * 256 + threadIdx.x;
  if (i < n) p[i] = 0.f;
}

// ---------------------------------------------------------------------------
// bf16 WMMA GEMM: C[M,N] = act(A[M,K] @ Bt[N,K]^T + bias), M=32768, N=K=1024
//
// block = 256 threads (8 waves, 4M x 2N). Block tile 128x128.
// Wave tile 32x64: 2 M-subtiles x 4 N-subtiles = 8 accumulators; each B
// fragment feeds two WMMAs. B is staged in double-buffered LDS (padded
// stride-40 rows -> conflict-free ds_load_b128) via the async global->LDS
// engine; next-step A fragments are prefetched into VGPRs so loads for k+32
// fly under the 8 WMMAs of step k.  act: 0 = none, 1 = ELU.
// ---------------------------------------------------------------------------
__global__ __launch_bounds__(256) void gemm_bf16_kernel(
    const __bf16* __restrict__ A, const __bf16* __restrict__ Bt,
    const float* __restrict__ bias, float* __restrict__ C, int act) {
  constexpr int Kd = HH, Nd = HH;
  constexpr int BSTR = 40;                          // padded row stride (bf16)
  alignas(16) __shared__ __bf16 Bs[2][128 * BSTR];  // 2 x 10 KB

  const int t     = threadIdx.x;
  const int wave  = t >> 5;
  const int lane  = t & 31;
  const int lrow  = lane & 15;
  const int lhi   = lane >> 4;
  const int waveM = wave >> 1;                      // 0..3
  const int waveN = wave & 1;                       // 0..1
  const int m0    = blockIdx.x * 128 + waveM * 32;
  const int nb0   = blockIdx.y * 128;
  const int n0    = nb0 + waveN * 64;

  const __bf16* Arow0 = A + (size_t)(m0 + lrow) * Kd;
  const __bf16* Arow1 = Arow0 + (size_t)16 * Kd;

  // staging assignment: 128 rows x 32 k / 256 threads = 16 bf16 (2 x b128) each
  const int snl = t >> 1;                           // 0..127 (local n)
  const int skl = (t & 1) * 16;                     // 0 or 16 (local k)
  const __bf16* Bsrc = Bt + (size_t)(nb0 + snl) * Kd + skl;
  __bf16* Bdst[2] = { &Bs[0][snl * BSTR + skl], &Bs[1][snl * BSTR + skl] };

  v8f acc[2][4];
#pragma unroll
  for (int mt = 0; mt < 2; ++mt)
#pragma unroll
    for (int tt = 0; tt < 4; ++tt) acc[mt][tt] = v8f{};

  // ---- prologue: stage k=0 into buffer 0, prefetch A(0) fragments
#if USE_ASYNC_LDS
  async_b128x2(Bsrc, Bdst[0]);
#else
  *(bf16x8*)Bdst[0]       = *(const bf16x8*)(Bsrc);
  *(bf16x8*)(Bdst[0] + 8) = *(const bf16x8*)(Bsrc + 8);
#endif
  BF16x16 a0n, a1n;
  a0n.h[0] = *(const bf16x8*)(Arow0 + lhi * 8);
  a0n.h[1] = *(const bf16x8*)(Arow0 + lhi * 8 + 16);
  a1n.h[0] = *(const bf16x8*)(Arow1 + lhi * 8);
  a1n.h[1] = *(const bf16x8*)(Arow1 + lhi * 8 + 16);
#if USE_ASYNC_LDS
  wait_async0();
#endif
  __syncthreads();

  for (int kk = 0; kk < Kd; kk += 32) {
    const int cur = (kk >> 5) & 1;
    const __bf16* BsCur = Bs[cur];
    __bf16* BdstN = Bdst[cur ^ 1];
    const bool more = (kk + 32) < Kd;

    const BF16x16 a0 = a0n, a1 = a1n;

    // issue next-step staging + A prefetch before the WMMAs
#if USE_ASYNC_LDS
    if (more) async_b128x2(Bsrc + kk + 32, BdstN);
#else
    bf16x8 st0, st1;
    if (more) {
      st0 = *(const bf16x8*)(Bsrc + kk + 32);
      st1 = *(const bf16x8*)(Bsrc + kk + 40);
    }
#endif
    if (more) {
      a0n.h[0] = *(const bf16x8*)(Arow0 + kk + 32 + lhi * 8);
      a0n.h[1] = *(const bf16x8*)(Arow0 + kk + 48 + lhi * 8);
      a1n.h[0] = *(const bf16x8*)(Arow1 + kk + 32 + lhi * 8);
      a1n.h[1] = *(const bf16x8*)(Arow1 + kk + 48 + lhi * 8);
    }

    // 8 WMMAs; each B fragment reused by the two M-subtiles
#pragma unroll
    for (int tt = 0; tt < 4; ++tt) {
      const int nl = waveN * 64 + tt * 16 + lrow;
      BF16x16 b;
      b.h[0] = *(const bf16x8*)&BsCur[nl * BSTR + lhi * 8];
      b.h[1] = *(const bf16x8*)&BsCur[nl * BSTR + lhi * 8 + 16];
      acc[0][tt] = __builtin_amdgcn_wmma_f32_16x16x32_bf16(
          false, a0.v, false, b.v, (short)0, acc[0][tt], false, false);
      acc[1][tt] = __builtin_amdgcn_wmma_f32_16x16x32_bf16(
          false, a1.v, false, b.v, (short)0, acc[1][tt], false, false);
    }

    if (more) {
#if USE_ASYNC_LDS
      wait_async0();
#else
      *(bf16x8*)BdstN       = st0;
      *(bf16x8*)(BdstN + 8) = st1;
#endif
    }
    __syncthreads();
  }

  // Epilogue: C/D layout — lane = column, VGPR r = row r (lanes>=16: row 8+r)
#pragma unroll
  for (int tt = 0; tt < 4; ++tt) {
    const int n = n0 + tt * 16 + lrow;
    const float bv_ = bias[n];
#pragma unroll
    for (int mt = 0; mt < 2; ++mt) {
#pragma unroll
      for (int r = 0; r < 8; ++r) {
        const int m = m0 + mt * 16 + lhi * 8 + r;
        float v = acc[mt][tt][r] + bv_;
        if (act) v = v > 0.f ? v : (__expf(v) - 1.f);
        C[(size_t)m * Nd + n] = v;
      }
    }
  }
}

// ---------------------------------------------------------------------------
// Per-(token,head) L2 normalize over DH=64, with optional extra scale
// (q gets scale = 1/sqrt(DH) folded in). One wave per 64-elem row.
// ---------------------------------------------------------------------------
__global__ __launch_bounds__(256) void l2norm_head_kernel(
    float* __restrict__ x, float scale, int nrows) {
  int wave = threadIdx.x >> 5, lane = threadIdx.x & 31;
  int row = blockIdx.x * 8 + wave;
  if (row >= nrows) return;
  float* p = x + (size_t)row * DHH;
  float a = p[lane], b = p[lane + 32];
  float ss = a * a + b * b;
#pragma unroll
  for (int off = 16; off > 0; off >>= 1) ss += __shfl_xor(ss, off, 32);
  float r = rsqrtf(ss) * scale;
  p[lane] = a * r;
  p[lane + 32] = b * r;
}

// ---------------------------------------------------------------------------
// kv[b,h,d,e] = sum_s kn[b,s,h,d] * v[b,s,h,e]   (atomic accumulation over
// S-chunks). grid = (128 head-pairs, 16 chunks of 256 s). Thread owns 16
// (d,e) outputs; kn/v tiles staged in LDS.
// ---------------------------------------------------------------------------
__global__ __launch_bounds__(256) void kv_kernel(
    const float* __restrict__ Kb, const float* __restrict__ Vb,
    float* __restrict__ kv) {
  alignas(16) __shared__ float ks[32][64];
  alignas(16) __shared__ float vs[32][64];
  const int t = threadIdx.x;
  const int p = blockIdx.x;                 // 0..127  (b*16 + h)
  const int b = p >> 4, h = p & 15;
  const int s0 = blockIdx.y * 256;
  const int d = t >> 2, e0 = (t & 3) * 16;

  float acc[16];
#pragma unroll
  for (int i = 0; i < 16; ++i) acc[i] = 0.f;

  const int sl = t >> 3, col = (t & 7) * 8;
  const size_t headOff = (size_t)b * SS * HH + (size_t)h * DHH;

  for (int sb = 0; sb < 8; ++sb) {
    const size_t rowOff = headOff + (size_t)(s0 + sb * 32 + sl) * HH + col;
    *(f32x4*)&ks[sl][col]     = *(const f32x4*)(Kb + rowOff);
    *(f32x4*)&ks[sl][col + 4] = *(const f32x4*)(Kb + rowOff + 4);
    *(f32x4*)&vs[sl][col]     = *(const f32x4*)(Vb + rowOff);
    *(f32x4*)&vs[sl][col + 4] = *(const f32x4*)(Vb + rowOff + 4);
    __syncthreads();
    for (int s = 0; s < 32; ++s) {
      const float kd = ks[s][d];
#pragma unroll
      for (int i = 0; i < 16; ++i) acc[i] += kd * vs[s][e0 + i];
    }
    __syncthreads();
  }
  float* outp = kv + (size_t)p * (DHH * DHH) + d * DHH + e0;
#pragma unroll
  for (int i = 0; i < 16; ++i) atomicAdd(outp + i, acc[i]);
}

// ---------------------------------------------------------------------------
// ctx[b,s,h,e] = sum_d qn[b,s,h,d] * kv[b,h,d,e]  (qn pre-scaled by 1/8).
// Writes bf16 directly (consumed by the output-projection WMMA GEMM).
// grid = (128 head-pairs, 128 chunks of 32 s). kv tile staged in LDS.
// ---------------------------------------------------------------------------
__global__ __launch_bounds__(256) void ctx_kernel(
    const float* __restrict__ Qb, const float* __restrict__ kv,
    __bf16* __restrict__ ctx) {
  alignas(16) __shared__ float kvs[64 * 64];      // 16 KB
  __shared__ float qs[8][64];
  const int t = threadIdx.x, w = t >> 5, lane = t & 31;
  const int p = blockIdx.x, b = p >> 4, h = p & 15;
  const int s0 = blockIdx.y * 32;

  const float* kvp = kv + (size_t)p * (DHH * DHH) + t * 16;
#pragma unroll
  for (int i = 0; i < 16; i += 4)
    *(f32x4*)&kvs[t * 16 + i] = *(const f32x4*)(kvp + i);
  __syncthreads();

  const size_t headOff = (size_t)b * SS * HH + (size_t)h * DHH;
  for (int j = 0; j < 4; ++j) {                   // uniform across all waves
    const int s = s0 + w * 4 + j;
    const float* qp = Qb + headOff + (size_t)s * HH;
    qs[w][lane] = qp[lane];
    qs[w][lane + 32] = qp[lane + 32];
    __syncthreads();
    float a0 = 0.f, a1 = 0.f;
#pragma unroll 8
    for (int d = 0; d < 64; ++d) {
      const float qd = qs[w][d];
      a0 += qd * kvs[d * 64 + lane];
      a1 += qd * kvs[d * 64 + lane + 32];
    }
    __bf16* op = ctx + headOff + (size_t)s * HH;
    op[lane] = f2bf(a0);
    op[lane + 32] = f2bf(a1);
    __syncthreads();
  }
}

// ---------------------------------------------------------------------------
// Fused residual + LayerNorm: out = LN(hidden + input) * gamma + beta
// One 256-thread block per 1024-wide row.
// ---------------------------------------------------------------------------
__global__ __launch_bounds__(256) void layernorm_kernel(
    const float* __restrict__ hid, const float* __restrict__ inp,
    const float* __restrict__ gamma, const float* __restrict__ beta,
    float* __restrict__ out) {
  __shared__ float r1[8], r2[8];
  const int t = threadIdx.x, w = t >> 5, lane = t & 31;
  const size_t base = (size_t)blockIdx.x * HH + t * 4;

  f32x4 hv = *(const f32x4*)(hid + base);
  f32x4 xv = *(const f32x4*)(inp + base);
  float x0 = hv[0] + xv[0], x1 = hv[1] + xv[1];
  float x2 = hv[2] + xv[2], x3 = hv[3] + xv[3];
  float s1 = x0 + x1 + x2 + x3;
  float s2 = x0 * x0 + x1 * x1 + x2 * x2 + x3 * x3;
#pragma unroll
  for (int off = 16; off > 0; off >>= 1) {
    s1 += __shfl_xor(s1, off, 32);
    s2 += __shfl_xor(s2, off, 32);
  }
  if (lane == 0) { r1[w] = s1; r2[w] = s2; }
  __syncthreads();
  float ts1 = 0.f, ts2 = 0.f;
#pragma unroll
  for (int k = 0; k < 8; ++k) { ts1 += r1[k]; ts2 += r2[k]; }
  const float mu = ts1 * (1.f / HH);
  const float var = ts2 * (1.f / HH) - mu * mu;
  const float rstd = rsqrtf(var + 1e-12f);

  f32x4 gv = *(const f32x4*)(gamma + t * 4);
  f32x4 bv = *(const f32x4*)(beta + t * 4);
  f32x4 ov;
  ov[0] = (x0 - mu) * rstd * gv[0] + bv[0];
  ov[1] = (x1 - mu) * rstd * gv[1] + bv[1];
  ov[2] = (x2 - mu) * rstd * gv[2] + bv[2];
  ov[3] = (x3 - mu) * rstd * gv[3] + bv[3];
  *(f32x4*)(out + base) = ov;
}

// ---------------------------------------------------------------------------
// Orchestration
// ---------------------------------------------------------------------------
extern "C" void kernel_launch(void* const* d_in, const int* in_sizes, int n_in,
                              void* d_out, int out_size, void* d_ws, size_t ws_size,
                              hipStream_t stream) {
  (void)in_sizes; (void)n_in; (void)out_size; (void)ws_size;

  const float* X     = (const float*)d_in[0];
  const float* Wq    = (const float*)d_in[1];
  const float* bq    = (const float*)d_in[2];
  const float* Wk    = (const float*)d_in[3];
  const float* bk    = (const float*)d_in[4];
  const float* Wv    = (const float*)d_in[5];
  const float* bv    = (const float*)d_in[6];
  const float* Wd    = (const float*)d_in[7];
  const float* bd    = (const float*)d_in[8];
  const float* gamma = (const float*)d_in[9];
  const float* beta  = (const float*)d_in[10];
  float* out = (float*)d_out;
  char* ws = (char*)d_ws;

  // workspace layout (bytes)
  const size_t SZ_XB = (size_t)MM * HH * 2;       //  64 MB  bf16 X
  const size_t SZ_WT = (size_t)HH * HH * 2;       //   2 MB  per bf16 W^T
  const size_t SZ_F  = (size_t)MM * HH * 4;       // 128 MB  fp32 activations
  __bf16* Xb  = (__bf16*)(ws);
  __bf16* Wt0 = (__bf16*)(ws + SZ_XB);
  __bf16* Wt1 = Wt0 + (size_t)HH * HH;
  __bf16* Wt2 = Wt1 + (size_t)HH * HH;
  __bf16* Wt3 = Wt2 + (size_t)HH * HH;
  float*  Qb  = (float*)(ws + SZ_XB + 4 * SZ_WT);
  float*  Kb  = (float*)((char*)Qb + SZ_F);
  float*  Vb  = (float*)((char*)Kb + SZ_F);
  float*  kvb = (float*)((char*)Vb + SZ_F);       // 128*64*64 fp32 = 2 MB
  __bf16* ctxb = (__bf16*)Kb;                     // reuse: K dead after kv
  float*  hid  = Vb;                              // reuse: V dead after kv/ctx

  // 1) convert inputs to the bf16 WMMA path (L2-resident working set)
  cvt_f32_bf16_kernel<<<(MM * HH / 4 + 255) / 256, 256, 0, stream>>>(X, Xb, MM * HH / 4);
  transpose_w_kernel<<<dim3(32, 32), 256, 0, stream>>>(Wq, Wt0);
  transpose_w_kernel<<<dim3(32, 32), 256, 0, stream>>>(Wk, Wt1);
  transpose_w_kernel<<<dim3(32, 32), 256, 0, stream>>>(Wv, Wt2);
  transpose_w_kernel<<<dim3(32, 32), 256, 0, stream>>>(Wd, Wt3);
  zero_f32_kernel<<<(BB * NHH * DHH * DHH + 255) / 256, 256, 0, stream>>>(
      kvb, BB * NHH * DHH * DHH);

  // 2) projections (fused bias + ELU for q,k)
  dim3 ggrid(MM / 128, HH / 128);
  gemm_bf16_kernel<<<ggrid, 256, 0, stream>>>(Xb, Wt0, bq, Qb, 1);
  gemm_bf16_kernel<<<ggrid, 256, 0, stream>>>(Xb, Wt1, bk, Kb, 1);
  gemm_bf16_kernel<<<ggrid, 256, 0, stream>>>(Xb, Wt2, bv, Vb, 0);

  // 3) L2 normalize per (token, head); fold 1/sqrt(DH) into q
  const int nrows = BB * SS * NHH;
  l2norm_head_kernel<<<nrows / 8, 256, 0, stream>>>(Qb, 0.125f, nrows);
  l2norm_head_kernel<<<nrows / 8, 256, 0, stream>>>(Kb, 1.0f, nrows);

  // 4) linear attention: kv = kn^T v ; ctx = qn kv (writes bf16)
  kv_kernel<<<dim3(BB * NHH, SS / 256), 256, 0, stream>>>(Kb, Vb, kvb);
  ctx_kernel<<<dim3(BB * NHH, SS / 32), 256, 0, stream>>>(Qb, kvb, ctxb);

  // 5) output projection + fused residual LayerNorm
  gemm_bf16_kernel<<<ggrid, 256, 0, stream>>>(ctxb, Wt3, bd, hid, 0);
  layernorm_kernel<<<MM, 256, 0, stream>>>(hid, X, gamma, beta, out);
}
